// MultiLayerNormLSTM_77481210020039
// MI455X (gfx1250) — matrix operand order repaired and checked by reading, not compiled, not run on hardware
//
#include <hip/hip_runtime.h>
#include <hip/hip_bf16.h>
#include <math.h>

// ---------------------------------------------------------------------------
// MultiLayerNorm-LSTM for MI455X (gfx1250), bf16 WMMA 16x16x32, f32 accumulate
// Weights pre-transposed to [N][K]; ping-pong LDS double buffering; all B
// fragments hoisted ahead of the WMMA burst so DS loads overlap matrix ops.
// ---------------------------------------------------------------------------

typedef __attribute__((ext_vector_type(16))) __bf16 v16bf;
typedef __attribute__((ext_vector_type(8)))  float  v8f;
typedef __attribute__((ext_vector_type(8)))  __bf16 bf16x8;
typedef __attribute__((ext_vector_type(4)))  __bf16 bf16x4;

#define HIDDEN   1024
#define GATES    4096      // 4*HIDDEN
#define INSZ     512
#define BATCH    128
#define SEQ      512
#define ACAT_LD  2560      // [ x_t (512) | h0 (1024) | h1 (1024) ]

static __device__ __forceinline__ __bf16 f2bf(float f) {
  union { float f; unsigned u; } v; v.f = f;
  unsigned r = (v.u + 0x7FFFu + ((v.u >> 16) & 1u)) >> 16;  // round-nearest-even
  unsigned short s = (unsigned short)r;
  __bf16 o; __builtin_memcpy(&o, &s, 2);
  return o;
}

// -------------------------------- utility kernels --------------------------

__global__ void zero_u32_kernel(unsigned* __restrict__ p, int n) {
  int i = blockIdx.x * blockDim.x + threadIdx.x;
  if (i < n) p[i] = 0u;
}

// Transpose-convert: in[K][N] f32 -> out[N][K] bf16.  K, N multiples of 32.
__global__ __launch_bounds__(256)
void transpose_f32_to_bf16_kernel(const float* __restrict__ in,
                                  __bf16* __restrict__ out, int K, int N) {
  __shared__ __bf16 tile[32][34];
  const int k0 = blockIdx.y * 32;
  const int n0 = blockIdx.x * 32;
  const int tx = threadIdx.x, ty = threadIdx.y;
#pragma unroll
  for (int r = ty; r < 32; r += 8)
    tile[r][tx] = f2bf(in[(size_t)(k0 + r) * N + (n0 + tx)]);
  __syncthreads();
#pragma unroll
  for (int r = ty; r < 32; r += 8)
    out[(size_t)(n0 + r) * K + (k0 + tx)] = tile[tx][r];
}

// Scatter x[:, t, :] (f32) into acat[:, 0:512] as bf16
__global__ void gather_x_kernel(const float* __restrict__ x,
                                __bf16* __restrict__ acat, int t) {
  int i = blockIdx.x * blockDim.x + threadIdx.x;
  if (i >= BATCH * INSZ) return;
  int b = i >> 9;           // /512
  int j = i & 511;
  acat[b * ACAT_LD + j] = f2bf(x[((size_t)b * SEQ + t) * INSZ + j]);
}

// -------------------------------- WMMA GEMM --------------------------------
// C[M,N] = A[M,K] @ B[K,N] + bias[N], with B supplied TRANSPOSED: Bt[N][K].
// Requires: M % 64 == 0, N % 128 == 0, K % 32 == 0.
#define BM 64
#define BN 128
#define BK 32
#define PAD 8

__global__ __launch_bounds__(256)
void gemm_bf16_wmma_kernel(const __bf16* __restrict__ A, int lda,
                           const __bf16* __restrict__ Bt, int ldbt,
                           float* __restrict__ C, int ldc,
                           const float* __restrict__ bias,
                           int M, int N, int K) {
  __shared__ __bf16 As [2][BM][BK + PAD];   // A tile, row-major [m][k]
  __shared__ __bf16 Bst[2][BN][BK + PAD];   // B tile, [n][k] (pre-transposed)

  const int tid  = threadIdx.x;
  const int lane = tid & 31;
  const int wid  = tid >> 5;       // 0..7
  const int wm   = wid >> 1;       // 0..3 : 16-row band within block
  const int wn   = wid & 1;        // 0..1 : 64-col band within block

  const int bm0 = blockIdx.y * BM;
  const int bn0 = blockIdx.x * BN;

  v8f acc[4] = {};                 // 4 x (16x16 f32) accumulators per wave

  // global->LDS staging mapping (all 8B/16B vector ops)
  const int arow = tid >> 3;           // 0..31 (rows arow, arow+32)
  const int acol = (tid & 7) * 4;      // 4 bf16 = 8B
  const int bn_  = tid >> 1;           // 0..127
  const int bk_  = (tid & 1) * 16;     // 16 bf16 = 2 x 16B

  const __bf16* Arow0 = A  + (size_t)(bm0 + arow) * lda  + acol;
  const __bf16* Arow1 = Arow0 + (size_t)32 * lda;
  const __bf16* Brow  = Bt + (size_t)(bn0 + bn_)  * ldbt + bk_;

  const int nk = K / BK;
  bf16x4 ra0, ra1; bf16x8 rb0, rb1;

  // prologue: tile 0 -> regs -> LDS[0]
  ra0 = *(const bf16x4*)(Arow0);
  ra1 = *(const bf16x4*)(Arow1);
  rb0 = *(const bf16x8*)(Brow);
  rb1 = *(const bf16x8*)(Brow + 8);
  *(bf16x4*)&As [0][arow     ][acol] = ra0;
  *(bf16x4*)&As [0][arow + 32][acol] = ra1;
  *(bf16x8*)&Bst[0][bn_][bk_    ]    = rb0;
  *(bf16x8*)&Bst[0][bn_][bk_ + 8]    = rb1;
  __syncthreads();

  for (int ki = 0; ki < nk; ++ki) {
    const int buf = ki & 1;
    const int kn  = (ki + 1) * BK;

    // stage k+1 tile into registers (overlaps with the WMMA burst below)
    if (ki + 1 < nk) {
      ra0 = *(const bf16x4*)(Arow0 + kn);
      ra1 = *(const bf16x4*)(Arow1 + kn);
      rb0 = *(const bf16x8*)(Brow  + kn);
      rb1 = *(const bf16x8*)(Brow  + kn + 8);
    }
    // warm near caches for tile k+2 (global_prefetch_b8, WGP scope)
    if (ki + 2 < nk) {
      __builtin_prefetch(Arow0 + kn + BK, 0, 3);
      __builtin_prefetch(Brow  + kn + BK, 0, 3);
    }

    // ---- A fragment (16x32): lanes 0-15 row m=l, K {0..7,16..23};
    //      lanes 16-31 K {8..15,24..31} ----
    const int am = wm * 16 + (lane & 15);
    const int kh = (lane >> 4) * 8;
    bf16x8 alo = *(const bf16x8*)&As[buf][am][kh];
    bf16x8 ahi = *(const bf16x8*)&As[buf][am][16 + kh];
    v16bf afrag;
#pragma unroll
    for (int i = 0; i < 8; ++i) { afrag[i] = alo[i]; afrag[i + 8] = ahi[i]; }

    // ---- load ALL four B fragments first, then burst 4 WMMAs ----
    // (lets the compiler use partial s_wait_dscnt thresholds so WMMA n
    //  overlaps the DS loads of fragments n+1..3)
    const int kg = (lane >> 4) * 8;   // B: lanes 0-15 hold K {0..7,16..23}, 16-31 hold {8..15,24..31}
    const int nb = wn * 64 + (lane & 15);
    v16bf bfrag[4];
#pragma unroll
    for (int tn = 0; tn < 4; ++tn) {
      bf16x8 b0 = *(const bf16x8*)&Bst[buf][nb + tn * 16][kg];
      bf16x8 b1 = *(const bf16x8*)&Bst[buf][nb + tn * 16][kg + 16];
#pragma unroll
      for (int i = 0; i < 8; ++i) { bfrag[tn][i] = b0[i]; bfrag[tn][i + 8] = b1[i]; }
    }
#pragma unroll
    for (int tn = 0; tn < 4; ++tn) {
      acc[tn] = __builtin_amdgcn_wmma_f32_16x16x32_bf16(
          false, afrag, false, bfrag[tn], (short)0, acc[tn], false, false);
    }

    // commit k+1 tile to the other buffer; one barrier per iteration
    if (ki + 1 < nk) {
      *(bf16x4*)&As [buf ^ 1][arow     ][acol] = ra0;
      *(bf16x4*)&As [buf ^ 1][arow + 32][acol] = ra1;
      *(bf16x8*)&Bst[buf ^ 1][bn_][bk_    ]    = rb0;
      *(bf16x8*)&Bst[buf ^ 1][bn_][bk_ + 8]    = rb1;
    }
    __syncthreads();
  }

  // ---- epilogue: C/D layout — lane = N, lane-group selects M half, VGPR r -> M=r(+8)
  const int rh = (lane >> 4) * 8;
  const int cn = lane & 15;
#pragma unroll
  for (int tn = 0; tn < 4; ++tn) {
    const int col = bn0 + wn * 64 + tn * 16 + cn;
    const float bv = bias ? bias[col] : 0.0f;
#pragma unroll
    for (int r = 0; r < 8; ++r) {
      const int row = bm0 + wm * 16 + rh + r;
      C[(size_t)row * ldc + col] = acc[tn][r] + bv;
    }
  }
}

// ---------------------------- LayerNorm + LSTM cell ------------------------
__global__ __launch_bounds__(256)
void lstm_cell_kernel(const float* __restrict__ gates,
                      const float* __restrict__ gamma,
                      const float* __restrict__ beta,
                      float* __restrict__ c_state,     // [128,1024]
                      __bf16* __restrict__ h_out) {    // acat slot, stride ACAT_LD
  const int b = blockIdx.x;
  const float* g = gates + (size_t)b * GATES;

  __shared__ float red1[256];
  __shared__ float red2[256];

  float s = 0.f, s2 = 0.f;
  for (int j = threadIdx.x; j < GATES; j += 256) {
    float v = g[j]; s += v; s2 += v * v;
  }
  red1[threadIdx.x] = s; red2[threadIdx.x] = s2;
  __syncthreads();
  for (int off = 128; off > 0; off >>= 1) {
    if (threadIdx.x < off) {
      red1[threadIdx.x] += red1[threadIdx.x + off];
      red2[threadIdx.x] += red2[threadIdx.x + off];
    }
    __syncthreads();
  }
  const float mean = red1[0] * (1.0f / GATES);
  const float var  = red2[0] * (1.0f / GATES) - mean * mean;
  const float rstd = rsqrtf(var + 1e-5f);

  for (int j = threadIdx.x; j < HIDDEN; j += 256) {
    const int ji = j, jf = HIDDEN + j, jg = 2 * HIDDEN + j, jo = 3 * HIDDEN + j;
    float zi = (g[ji] - mean) * rstd * gamma[ji] + beta[ji];
    float zf = (g[jf] - mean) * rstd * gamma[jf] + beta[jf];
    float zg = (g[jg] - mean) * rstd * gamma[jg] + beta[jg];
    float zo = (g[jo] - mean) * rstd * gamma[jo] + beta[jo];
    float ii = 1.f / (1.f + __expf(-zi));
    float ff = 1.f / (1.f + __expf(-zf));
    float gg = tanhf(zg);
    float oo = 1.f / (1.f + __expf(-zo));
    float c  = ff * c_state[(size_t)b * HIDDEN + j] + ii * gg;
    c_state[(size_t)b * HIDDEN + j] = c;
    float h  = oo * tanhf(c);
    h_out[(size_t)b * ACAT_LD + j] = f2bf(h);
  }
}

// -------------------------------- launcher ---------------------------------

extern "C" void kernel_launch(void* const* d_in, const int* in_sizes, int n_in,
                              void* d_out, int out_size, void* d_ws, size_t ws_size,
                              hipStream_t stream) {
  const float* x   = (const float*)d_in[0];
  const float* W0  = (const float*)d_in[1];
  const float* b0  = (const float*)d_in[2];
  const float* g0  = (const float*)d_in[3];
  const float* be0 = (const float*)d_in[4];
  const float* W1  = (const float*)d_in[5];
  const float* b1  = (const float*)d_in[6];
  const float* g1  = (const float*)d_in[7];
  const float* be1 = (const float*)d_in[8];
  const float* fcW = (const float*)d_in[9];
  const float* fcb = (const float*)d_in[10];
  float* out = (float*)d_out;

  // workspace layout (all offsets 256B aligned). Weights stored TRANSPOSED [N][K].
  char* ws = (char*)d_ws;
  __bf16* W0t  = (__bf16*)ws;  ws += (size_t)4096 * 1536 * 2;   // 12.6 MB
  __bf16* W1t  = (__bf16*)ws;  ws += (size_t)4096 * 2048 * 2;   // 16.8 MB
  __bf16* fcWt = (__bf16*)ws;  ws += (size_t)512  * 1024 * 2;   //  1.0 MB
  __bf16* acat = (__bf16*)ws;  ws += (size_t)BATCH * ACAT_LD * 2;
  float*  c0   = (float*)ws;   ws += (size_t)BATCH * HIDDEN * 4;
  float*  c1   = (float*)ws;   ws += (size_t)BATCH * HIDDEN * 4;
  float*  gates = (float*)ws;  ws += (size_t)BATCH * GATES  * 4;

  const dim3 blk(256);
  const dim3 tblk(32, 8);

  // 1) transpose-convert weights to bf16 [N][K] (L2-resident: ~30 MB << 192 MB)
  transpose_f32_to_bf16_kernel<<<dim3(4096 / 32, 1536 / 32), tblk, 0, stream>>>(
      W0, W0t, 1536, 4096);
  transpose_f32_to_bf16_kernel<<<dim3(4096 / 32, 2048 / 32), tblk, 0, stream>>>(
      W1, W1t, 2048, 4096);
  transpose_f32_to_bf16_kernel<<<dim3(512 / 32, 1024 / 32), tblk, 0, stream>>>(
      fcW, fcWt, 1024, 512);

  // 2) zero state: acat (bf16 zeros) + c0 + c1 are contiguous
  {
    int zwords = (BATCH * ACAT_LD * 2 + 2 * BATCH * HIDDEN * 4) / 4;
    zero_u32_kernel<<<(zwords + 255) / 256, blk, 0, stream>>>((unsigned*)acat, zwords);
  }

  const dim3 gemm_gates_grid(GATES / BN, BATCH / BM);  // 32 x 2
  const dim3 gemm_out_grid(512 / BN, BATCH / BM);      //  4 x 2

  // 3) sequential recurrence
  for (int t = 0; t < SEQ; ++t) {
    gather_x_kernel<<<(BATCH * INSZ + 255) / 256, blk, 0, stream>>>(x, acat, t);

    // layer 0: gates = [x_t | h0] @ W0 + b0   (A = acat[:,0:1536], lda = 2560)
    gemm_bf16_wmma_kernel<<<gemm_gates_grid, blk, 0, stream>>>(
        acat, ACAT_LD, W0t, 1536, gates, GATES, b0, BATCH, GATES, 1536);
    lstm_cell_kernel<<<BATCH, blk, 0, stream>>>(gates, g0, be0, c0, acat + 512);

    // layer 1: gates = [h0 | h1] @ W1 + b1    (A = acat[:,512:2560], lda = 2560)
    gemm_bf16_wmma_kernel<<<gemm_gates_grid, blk, 0, stream>>>(
        acat + 512, ACAT_LD, W1t, 2048, gates, GATES, b1, BATCH, GATES, 2048);
    lstm_cell_kernel<<<BATCH, blk, 0, stream>>>(gates, g1, be1, c1, acat + 1536);
  }

  // 4) final FC: out[128,512] = h1 @ fcW + fcb
  gemm_bf16_wmma_kernel<<<gemm_out_grid, blk, 0, stream>>>(
      acat + 1536, ACAT_LD, fcWt, 1024, out, 512, fcb, BATCH, 512, 1024);
}
